// PointNet2PureSemSeg_85066122265455
// MI455X (gfx1250) — compile-verified, matrix operand order checked by reading, not run on hardware
//
#include <hip/hip_runtime.h>
#include <hip/hip_bf16.h>

// ---------------------------------------------------------------------------
// PointNet++ semantic segmentation forward, CDNA5 (gfx1250, wave32, WMMA).
// All MLP matmuls run through v_wmma_f32_16x16x32_bf16. Weights are pre-packed
// once per launch into zero-padded bf16 (Cout x KP) so A and B fragments are
// each two 16-byte loads per lane (no predication in the WMMA inner loop).
// Set-Abstraction stages are fully fused (gather -> 3 WMMA layers -> maxpool)
// with LDS ping-pong activation buffers so grouped features never touch HBM.
// ---------------------------------------------------------------------------

typedef __attribute__((ext_vector_type(16))) __bf16 v16bf;
typedef __attribute__((ext_vector_type(8)))  __bf16 v8bf;
typedef __attribute__((ext_vector_type(8)))  float  v8f;

#define SA_STRIDE 520   // bf16 elements per LDS row (512 + 8 pad -> conflict-free)

// ---- WMMA fragment helper (layouts per CDNA5 ISA 7.12.2) -------
// Row-major bf16 tile fragment: lane lo = row (A) / output channel (B),
// K elems = [8*hi, 8*hi+8) and [16+8*hi, 24+8*hi). rowp = row base + k0.
// Works for both A (activations) and B (pre-packed weights, column-major
// mirror == row-major over (Cout, K)).
__device__ __forceinline__ v16bf load_frag(const __bf16* rowp, int hi) {
  v16bf a;
  v8bf p0 = *(const v8bf*)(rowp + 8 * hi);
  v8bf p1 = *(const v8bf*)(rowp + 16 + 8 * hi);
#pragma unroll
  for (int i = 0; i < 8; ++i) { a[i] = p0[i]; a[i + 8] = p1[i]; }
  return a;
}

// ---------------------------------------------------------------------------
// Pack fp32 weights (Cout x Cin, row-major) -> bf16 (Cout x KP), zero padded.
__global__ void pack_w_kernel(const float* __restrict__ W,
                              __bf16* __restrict__ out,
                              int Cin, int KP, int total) {
  int t = blockIdx.x * blockDim.x + threadIdx.x;
  if (t >= total) return;
  int o = t / KP, k = t % KP;
  out[t] = (k < Cin) ? (__bf16)W[(size_t)o * Cin + k] : (__bf16)0.0f;
}

// ---------------------------------------------------------------------------
// xyz (B,3,N) -> (B,N,3)
__global__ void transpose_xyz_kernel(const float* __restrict__ in,
                                     float* __restrict__ out, int BB, int NN) {
  int t = blockIdx.x * blockDim.x + threadIdx.x;
  if (t >= BB * NN) return;
  int b = t / NN, i = t % NN;
  out[(size_t)t * 3 + 0] = in[((size_t)b * 3 + 0) * NN + i];
  out[(size_t)t * 3 + 1] = in[((size_t)b * 3 + 1) * NN + i];
  out[(size_t)t * 3 + 2] = in[((size_t)b * 3 + 2) * NN + i];
}

// ---------------------------------------------------------------------------
// Farthest point sampling: one block per batch, dist array in LDS, serial
// npoint iterations with block-wide argmax (wave shfl + LDS reduction).
__global__ __launch_bounds__(1024) void fps_kernel(
    const float* __restrict__ xyz, int n, int npoint,
    int* __restrict__ out_idx, float* __restrict__ new_xyz) {
  __shared__ float dist[8192];
  __shared__ float rv[32];
  __shared__ int ri[32];
  __shared__ int far_s;
  int b = blockIdx.x, tid = threadIdx.x;
  const float* xb = xyz + (size_t)b * n * 3;
  for (int i = tid; i < n; i += blockDim.x) dist[i] = 1e10f;
  if (tid == 0) far_s = 0;
  __syncthreads();
  for (int it = 0; it < npoint; ++it) {
    int far = far_s;
    if (tid == 0) out_idx[b * npoint + it] = far;
    if (tid < 3) new_xyz[((size_t)b * npoint + it) * 3 + tid] = xb[(size_t)far * 3 + tid];
    float cx = xb[(size_t)far * 3 + 0];
    float cy = xb[(size_t)far * 3 + 1];
    float cz = xb[(size_t)far * 3 + 2];
    float bestv = -1.0f; int besti = 0;
    for (int i = tid; i < n; i += blockDim.x) {
      float dx = xb[(size_t)i * 3 + 0] - cx;
      float dy = xb[(size_t)i * 3 + 1] - cy;
      float dz = xb[(size_t)i * 3 + 2] - cz;
      float d = fminf(dist[i], dx * dx + dy * dy + dz * dz);
      dist[i] = d;
      if (d > bestv) { bestv = d; besti = i; }
    }
#pragma unroll
    for (int off = 16; off > 0; off >>= 1) {
      float ov = __shfl_down(bestv, off, 32);
      int   oi = __shfl_down(besti, off, 32);
      if (ov > bestv || (ov == bestv && oi < besti)) { bestv = ov; besti = oi; }
    }
    int wid = tid >> 5;
    if ((tid & 31) == 0) { rv[wid] = bestv; ri[wid] = besti; }
    __syncthreads();
    if (tid == 0) {
      float bv = rv[0]; int bi = ri[0];
      int nw = blockDim.x >> 5;
      for (int w = 1; w < nw; ++w)
        if (rv[w] > bv || (rv[w] == bv && ri[w] < bi)) { bv = rv[w]; bi = ri[w]; }
      far_s = bi;
    }
    __syncthreads();
  }
}

// ---------------------------------------------------------------------------
// Brute-force 32-NN: per-thread candidate set in LDS (avoids scratch spill
// from dynamic register indexing).
#define KNN_T 128
__global__ __launch_bounds__(KNN_T) void knn_kernel(
    const float* __restrict__ q, const float* __restrict__ src,
    int S, int n, int BB, int* __restrict__ idx) {
  __shared__ float bd[32 * KNN_T];
  __shared__ int   bi[32 * KNN_T];
  int tid = threadIdx.x;
  int t = blockIdx.x * KNN_T + tid;
  bool act = t < BB * S;
  int b = act ? t / S : 0, s = act ? t % S : 0;
  float x = 0.f, y = 0.f, z = 0.f;
  if (act) {
    const float* qp = q + ((size_t)b * S + s) * 3;
    x = qp[0]; y = qp[1]; z = qp[2];
  }
#pragma unroll
  for (int k = 0; k < 32; ++k) { bd[k * KNN_T + tid] = 1e30f; bi[k * KNN_T + tid] = 0; }
  if (act) {
    float worst = 1e30f; int wslot = 0;
    const float* sb = src + (size_t)b * n * 3;
    for (int p = 0; p < n; ++p) {
      float dx = sb[(size_t)p * 3 + 0] - x;
      float dy = sb[(size_t)p * 3 + 1] - y;
      float dz = sb[(size_t)p * 3 + 2] - z;
      float d = dx * dx + dy * dy + dz * dz;
      if (d < worst) {
        bd[wslot * KNN_T + tid] = d; bi[wslot * KNN_T + tid] = p;
        float w = -1.0f; int wi = 0;
#pragma unroll
        for (int k = 0; k < 32; ++k) {
          float v = bd[k * KNN_T + tid];
          if (v > w) { w = v; wi = k; }
        }
        worst = w; wslot = wi;
      }
    }
    int* op = idx + ((size_t)b * S + s) * 32;
#pragma unroll
    for (int k = 0; k < 32; ++k) op[k] = bi[k * KNN_T + tid];
  }
}

// ---------------------------------------------------------------------------
// Fused Set Abstraction: one block per (b, sampled point). 32 neighbor rows,
// 3 WMMA MLP layers through LDS ping-pong, then channel-wise max over k.
// Wp = bf16 pre-packed weights (Cout x K), K multiple of 32.
__device__ __forceinline__ void sa_layer(const __bf16* in, __bf16* out,
                                         const __bf16* Wp, const float* g,
                                         const float* bta, int K,
                                         int Cout, int wave, int lane) {
  int lo = lane & 15, hi = lane >> 4;
  int colTiles = Cout >> 4;
  for (int t = wave; t < 2 * colTiles; t += 8) {
    int rt = t & 1, ct = t >> 1;
    v8f acc = {};
    const __bf16* arow = in + (rt * 16 + lo) * SA_STRIDE;
    const __bf16* brow = Wp + (size_t)(ct * 16 + lo) * K;
    for (int k0 = 0; k0 < K; k0 += 32) {
      v16bf a = load_frag(arow + k0, hi);
      v16bf bb = load_frag(brow + k0, hi);
      acc = __builtin_amdgcn_wmma_f32_16x16x32_bf16(false, a, false, bb,
                                                    (short)0, acc, false, false);
    }
    int o = ct * 16 + lo;
    float gg = g[o], be = bta[o];
#pragma unroll
    for (int r = 0; r < 8; ++r) {
      float v = acc[r] * gg + be;
      v = v > 0.f ? v : 0.f;
      out[(rt * 16 + r + 8 * hi) * SA_STRIDE + o] = (__bf16)v;
    }
  }
}

__global__ __launch_bounds__(256) void sa_fused_kernel(
    const float* __restrict__ xyz_src,   // (B,n,3)
    const float* __restrict__ feats_src, // (B,n,Cf) or null
    const float* __restrict__ new_xyz,   // (B,S,3)
    const int* __restrict__ knn_idx,     // (B,S,32)
    const __bf16* W1, const float* g1, const float* b1,
    const __bf16* W2, const float* g2, const float* b2,
    const __bf16* W3, const float* g3, const float* b3,
    int n, int S, int Cf, int C1, int C2, int C3,
    float* __restrict__ out)             // (B,S,C3)
{
  __shared__ __align__(16) __bf16 bufA[32 * SA_STRIDE];
  __shared__ __align__(16) __bf16 bufB[32 * SA_STRIDE];
  __shared__ int sIdx[32];
  __shared__ float sNew[3];

  int bs = blockIdx.x;           // b*S + s
  int b = bs / S, s = bs % S;
  (void)s;
  int tid = threadIdx.x;
  int wave = tid >> 5, lane = tid & 31;
  int Cin = 3 + Cf;
  int K0P = (Cin + 31) & ~31;

  if (tid < 32) sIdx[tid] = knn_idx[(size_t)bs * 32 + tid];
  if (tid < 3)  sNew[tid] = new_xyz[(size_t)bs * 3 + tid];
  __syncthreads();

  // gather grouped features (local coords + point feats), zero-padded to K0P
  for (int t = tid; t < 32 * K0P; t += 256) {
    int j = t / K0P, c = t % K0P;
    float v = 0.f;
    int p = sIdx[j];
    if (c < 3)        v = xyz_src[((size_t)b * n + p) * 3 + c] - sNew[c];
    else if (c < Cin) v = feats_src[((size_t)b * n + p) * Cf + (c - 3)];
    bufA[j * SA_STRIDE + c] = (__bf16)v;
  }
  __syncthreads();

  sa_layer(bufA, bufB, W1, g1, b1, K0P, C1, wave, lane);
  __syncthreads();
  sa_layer(bufB, bufA, W2, g2, b2, C1, C2, wave, lane);
  __syncthreads();
  sa_layer(bufA, bufB, W3, g3, b3, C2, C3, wave, lane);
  __syncthreads();

  // max over the 32 neighbors
  for (int c = tid; c < C3; c += 256) {
    float m = -1e30f;
#pragma unroll 4
    for (int j = 0; j < 32; ++j) m = fmaxf(m, (float)bufB[j * SA_STRIDE + c]);
    out[(size_t)bs * C3 + c] = m;
  }
}

// ---------------------------------------------------------------------------
// Generic GEMM + BN-fold + optional ReLU: rows x KP (bf16) times Wp^T
// (Nc x KP bf16, pre-packed). One 16x16 output tile per wave, grid-stride.
__global__ __launch_bounds__(256) void gemm_bf16_kernel(
    const __bf16* __restrict__ A, int KP,
    const __bf16* __restrict__ Wp, const float* __restrict__ g,
    const float* __restrict__ bta,
    __bf16* __restrict__ outB, float* __restrict__ outF,
    int rows, int Nc, int relu) {
  int wave = (blockIdx.x * blockDim.x + threadIdx.x) >> 5;
  int lane = threadIdx.x & 31;
  int lo = lane & 15, hi = lane >> 4;
  int rowTiles = rows >> 4, colTiles = Nc >> 4;
  int total = rowTiles * colTiles;
  int stride = (gridDim.x * blockDim.x) >> 5;
  for (int t = wave; t < total; t += stride) {
    int rt = t % rowTiles, ct = t / rowTiles;
    int r0 = rt << 4, o0 = ct << 4;
    v8f acc = {};
    const __bf16* arow = A + (size_t)(r0 + lo) * KP;
    const __bf16* brow = Wp + (size_t)(o0 + lo) * KP;
    for (int k0 = 0; k0 < KP; k0 += 32) {
      v16bf a = load_frag(arow + k0, hi);
      v16bf bb = load_frag(brow + k0, hi);
      acc = __builtin_amdgcn_wmma_f32_16x16x32_bf16(false, a, false, bb,
                                                    (short)0, acc, false, false);
    }
    int o = o0 + lo;
    float gg = g[o], be = bta[o];
#pragma unroll
    for (int r = 0; r < 8; ++r) {
      float v = acc[r] * gg + be;
      if (relu) v = v > 0.f ? v : 0.f;
      size_t row = (size_t)(r0 + r + 8 * hi);
      if (outB) outB[row * Nc + o] = (__bf16)v;
      if (outF) outF[row * Nc + o] = v;
    }
  }
}

// ---------------------------------------------------------------------------
// FP: 3-NN inverse-distance interpolation, writes bf16 straight into the
// GEMM input buffer at column offset colOff.
__global__ void interp3_kernel(const float* __restrict__ xyz1,
                               const float* __restrict__ xyz2,
                               const float* __restrict__ pts2,
                               int n1, int n2, int C2,
                               __bf16* __restrict__ outA, int KP, int colOff,
                               int BB) {
  int t = blockIdx.x * blockDim.x + threadIdx.x;
  if (t >= BB * n1) return;
  int b = t / n1;
  const float* p1 = xyz1 + (size_t)t * 3;
  float x = p1[0], y = p1[1], z = p1[2];
  float d0 = 1e30f, d1 = 1e30f, d2 = 1e30f;
  int i0 = 0, i1 = 0, i2 = 0;
  const float* src = xyz2 + (size_t)b * n2 * 3;
  for (int p = 0; p < n2; ++p) {
    float dx = src[(size_t)p * 3 + 0] - x;
    float dy = src[(size_t)p * 3 + 1] - y;
    float dz = src[(size_t)p * 3 + 2] - z;
    float d = dx * dx + dy * dy + dz * dz;
    if (d < d0)      { d2 = d1; i2 = i1; d1 = d0; i1 = i0; d0 = d; i0 = p; }
    else if (d < d1) { d2 = d1; i2 = i1; d1 = d;  i1 = p; }
    else if (d < d2) { d2 = d;  i2 = p; }
  }
  d0 = fmaxf(d0, 1e-10f); d1 = fmaxf(d1, 1e-10f); d2 = fmaxf(d2, 1e-10f);
  float w0 = 1.f / d0, w1 = 1.f / d1, w2 = 1.f / d2;
  float ws = w0 + w1 + w2;
  w0 /= ws; w1 /= ws; w2 /= ws;
  const float* f0 = pts2 + ((size_t)b * n2 + i0) * C2;
  const float* f1 = pts2 + ((size_t)b * n2 + i1) * C2;
  const float* f2 = pts2 + ((size_t)b * n2 + i2) * C2;
  __bf16* dst = outA + (size_t)t * KP + colOff;
  for (int c = 0; c < C2; ++c)
    dst[c] = (__bf16)(w0 * f0[c] + w1 * f1[c] + w2 * f2[c]);
}

// copy fp32 skip features into bf16 GEMM input buffer (concat part 1)
__global__ void copy_pts_kernel(const float* __restrict__ pts,
                                __bf16* __restrict__ outA,
                                int C, int KP, int colOff, int total) {
  int t = blockIdx.x * blockDim.x + threadIdx.x;
  if (t >= total) return;
  int row = t / C, c = t % C;
  outA[(size_t)row * KP + colOff + c] = (__bf16)pts[t];
}

// final classifier: 128 -> 2, output layout (B, 2, N)
__global__ void cls2_kernel(const __bf16* __restrict__ h,
                            const float* __restrict__ w2,
                            const float* __restrict__ b2,
                            float* __restrict__ out, int BB, int NN) {
  int t = blockIdx.x * blockDim.x + threadIdx.x;
  if (t >= BB * NN) return;
  int b = t / NN, i = t % NN;
  const v8bf* hp = (const v8bf*)(h + (size_t)t * 128);
  float a0 = b2[0], a1 = b2[1];
#pragma unroll
  for (int v = 0; v < 16; ++v) {
    v8bf hv = hp[v];
#pragma unroll
    for (int e = 0; e < 8; ++e) {
      float x = (float)hv[e];
      int c = v * 8 + e;
      a0 += x * w2[c];
      a1 += x * w2[128 + c];
    }
  }
  out[((size_t)b * 2 + 0) * NN + i] = a0;
  out[((size_t)b * 2 + 1) * NN + i] = a1;
}

// ---------------------------------------------------------------------------
extern "C" void kernel_launch(void* const* d_in, const int* in_sizes, int n_in,
                              void* d_out, int out_size, void* d_ws, size_t ws_size,
                              hipStream_t stream) {
  (void)out_size; (void)ws_size;
  const int Bb = 8, Nn = 8192;

  // Input index maps: detect jax tree_leaves key-sorted vs insertion order.
  int I_xyz, I_sa1, I_sa2, I_sa3, I_fp3, I_fp2, I_fp1, I_cls1, I_c2w, I_c2b;
  if (n_in >= 2 && in_sizes[0] == Bb * 3 * Nn) {   // insertion order
    I_xyz = 0;  I_sa1 = 1;  I_sa2 = 10; I_sa3 = 19;
    I_fp3 = 28; I_fp2 = 34; I_fp1 = 40; I_cls1 = 49;
    I_c2w = 52; I_c2b = 53;
  } else {                                          // sorted dict keys
    I_cls1 = 0; I_c2b = 3;  I_c2w = 4;  I_fp1 = 5;
    I_fp2 = 14; I_fp3 = 20; I_sa1 = 26; I_sa2 = 35;
    I_sa3 = 44; I_xyz = 53;
  }
  auto F = [&](int i) { return (const float*)d_in[i]; };

  // Workspace carve-out
  char* wsb = (char*)d_ws;
  size_t off = 0;
  auto alloc = [&](size_t bytes) {
    void* p = wsb + off;
    off = (off + bytes + 255) & ~(size_t)255;
    return p;
  };
  float* l0   = (float*)alloc((size_t)Bb * Nn * 3 * 4);
  int*   fi1  = (int*)  alloc((size_t)Bb * 2048 * 4);
  float* nx1  = (float*)alloc((size_t)Bb * 2048 * 3 * 4);
  int*   knn1 = (int*)  alloc((size_t)Bb * 2048 * 32 * 4);
  float* l1p  = (float*)alloc((size_t)Bb * 2048 * 128 * 4);
  int*   fi2  = (int*)  alloc((size_t)Bb * 512 * 4);
  float* nx2  = (float*)alloc((size_t)Bb * 512 * 3 * 4);
  int*   knn2 = (int*)  alloc((size_t)Bb * 512 * 32 * 4);
  float* l2p  = (float*)alloc((size_t)Bb * 512 * 256 * 4);
  int*   fi3  = (int*)  alloc((size_t)Bb * 128 * 4);
  float* nx3  = (float*)alloc((size_t)Bb * 128 * 3 * 4);
  int*   knn3 = (int*)  alloc((size_t)Bb * 128 * 32 * 4);
  float* l3p  = (float*)alloc((size_t)Bb * 128 * 512 * 4);
  float* l2f  = (float*)alloc((size_t)Bb * 512 * 256 * 4);
  float* l1f  = (float*)alloc((size_t)Bb * 2048 * 128 * 4);
  __bf16* bufA = (__bf16*)alloc((size_t)Bb * Nn * 128 * 2);   // max act buffer
  __bf16* bufB = (__bf16*)alloc((size_t)Bb * Nn * 128 * 2);

  // Pre-pack all MLP weights to zero-padded bf16 (Cout x KP).
  // idx of fp32 W leaf, Cout, Cin, KP
  struct WInfo { int idx, cout, cin, kp; };
  const WInfo winfo[13] = {
    {I_sa1 + 0,  64,   3,  32}, {I_sa1 + 3,  64,  64,  64}, {I_sa1 + 6, 128,  64,  64},
    {I_sa2 + 0, 128, 131, 160}, {I_sa2 + 3, 128, 128, 128}, {I_sa2 + 6, 256, 128, 128},
    {I_sa3 + 0, 256, 259, 288}, {I_sa3 + 3, 256, 256, 256}, {I_sa3 + 6, 512, 256, 256},
    {I_fp3 + 0, 256, 768, 768}, {I_fp3 + 3, 256, 256, 256},
    {I_fp2 + 0, 256, 384, 384}, {I_fp2 + 3, 128, 256, 256},
  };
  __bf16* Wp[17];
  for (int w = 0; w < 13; ++w) {
    int total = winfo[w].cout * winfo[w].kp;
    Wp[w] = (__bf16*)alloc((size_t)total * 2);
    pack_w_kernel<<<(total + 255) / 256, 256, 0, stream>>>(
        F(winfo[w].idx), Wp[w], winfo[w].cin, winfo[w].kp, total);
  }
  // fp1 (3 layers) + cls1: all 128x128
  for (int l = 0; l < 4; ++l) {
    int idx = (l < 3) ? (I_fp1 + 3 * l) : I_cls1;
    int total = 128 * 128;
    Wp[13 + l] = (__bf16*)alloc((size_t)total * 2);
    pack_w_kernel<<<(total + 255) / 256, 256, 0, stream>>>(
        F(idx), Wp[13 + l], 128, 128, total);
  }

  auto gemm = [&](const __bf16* A, int KP, const __bf16* W, const float* g,
                  const float* bta, __bf16* oB, float* oF, int rows, int Nc,
                  int relu) {
    int total = (rows / 16) * (Nc / 16);
    int grid = (total + 7) / 8;
    if (grid > 16384) grid = 16384;
    gemm_bf16_kernel<<<grid, 256, 0, stream>>>(A, KP, W, g, bta, oB, oF,
                                               rows, Nc, relu);
  };

  // --- l0: transpose xyz ---
  transpose_xyz_kernel<<<(Bb * Nn + 255) / 256, 256, 0, stream>>>(
      F(I_xyz), l0, Bb, Nn);

  // --- SA1: 8192 -> 2048 pts, MLP 3->64->64->128 ---
  fps_kernel<<<Bb, 1024, 0, stream>>>(l0, Nn, 2048, fi1, nx1);
  knn_kernel<<<(Bb * 2048 + KNN_T - 1) / KNN_T, KNN_T, 0, stream>>>(
      nx1, l0, 2048, Nn, Bb, knn1);
  sa_fused_kernel<<<Bb * 2048, 256, 0, stream>>>(
      l0, nullptr, nx1, knn1,
      Wp[0], F(I_sa1 + 1), F(I_sa1 + 2),
      Wp[1], F(I_sa1 + 4), F(I_sa1 + 5),
      Wp[2], F(I_sa1 + 7), F(I_sa1 + 8),
      Nn, 2048, 0, 64, 64, 128, l1p);

  // --- SA2: 2048 -> 512 pts, MLP 131->128->128->256 ---
  fps_kernel<<<Bb, 1024, 0, stream>>>(nx1, 2048, 512, fi2, nx2);
  knn_kernel<<<(Bb * 512 + KNN_T - 1) / KNN_T, KNN_T, 0, stream>>>(
      nx2, nx1, 512, 2048, Bb, knn2);
  sa_fused_kernel<<<Bb * 512, 256, 0, stream>>>(
      nx1, l1p, nx2, knn2,
      Wp[3], F(I_sa2 + 1), F(I_sa2 + 2),
      Wp[4], F(I_sa2 + 4), F(I_sa2 + 5),
      Wp[5], F(I_sa2 + 7), F(I_sa2 + 8),
      2048, 512, 128, 128, 128, 256, l2p);

  // --- SA3: 512 -> 128 pts, MLP 259->256->256->512 ---
  fps_kernel<<<Bb, 1024, 0, stream>>>(nx2, 512, 128, fi3, nx3);
  knn_kernel<<<(Bb * 128 + KNN_T - 1) / KNN_T, KNN_T, 0, stream>>>(
      nx3, nx2, 128, 512, Bb, knn3);
  sa_fused_kernel<<<Bb * 128, 256, 0, stream>>>(
      nx2, l2p, nx3, knn3,
      Wp[6], F(I_sa3 + 1), F(I_sa3 + 2),
      Wp[7], F(I_sa3 + 4), F(I_sa3 + 5),
      Wp[8], F(I_sa3 + 7), F(I_sa3 + 8),
      512, 128, 256, 256, 256, 512, l3p);

  // --- FP3: interp l3 -> l2 points, concat(l2p,interp)=768 -> 256 -> 256 ---
  {
    int rows = Bb * 512, KP = 768;
    copy_pts_kernel<<<(rows * 256 + 255) / 256, 256, 0, stream>>>(
        l2p, bufA, 256, KP, 0, rows * 256);
    interp3_kernel<<<(rows + 255) / 256, 256, 0, stream>>>(
        nx2, nx3, l3p, 512, 128, 512, bufA, KP, 256, Bb);
    gemm(bufA, KP,  Wp[9],  F(I_fp3 + 1), F(I_fp3 + 2), bufB, nullptr, rows, 256, 1);
    gemm(bufB, 256, Wp[10], F(I_fp3 + 4), F(I_fp3 + 5), nullptr, l2f,  rows, 256, 1);
  }

  // --- FP2: interp l2f -> l1 points, concat(l1p,interp)=384 -> 256 -> 128 ---
  {
    int rows = Bb * 2048, KP = 384;
    copy_pts_kernel<<<(rows * 128 + 255) / 256, 256, 0, stream>>>(
        l1p, bufA, 128, KP, 0, rows * 128);
    interp3_kernel<<<(rows + 255) / 256, 256, 0, stream>>>(
        nx1, nx2, l2f, 2048, 512, 256, bufA, KP, 128, Bb);
    gemm(bufA, KP,  Wp[11], F(I_fp2 + 1), F(I_fp2 + 2), bufB, nullptr, rows, 256, 1);
    gemm(bufB, 256, Wp[12], F(I_fp2 + 4), F(I_fp2 + 5), nullptr, l1f,  rows, 128, 1);
  }

  // --- FP1: interp l1f -> l0 points (no skip), 128 -> 128 x3, cls1, cls2 ---
  {
    int rows = Bb * Nn;
    interp3_kernel<<<(rows + 255) / 256, 256, 0, stream>>>(
        l0, nx1, l1f, Nn, 2048, 128, bufA, 128, 0, Bb);
    gemm(bufA, 128, Wp[13], F(I_fp1 + 1), F(I_fp1 + 2), bufB, nullptr, rows, 128, 1);
    gemm(bufB, 128, Wp[14], F(I_fp1 + 4), F(I_fp1 + 5), bufA, nullptr, rows, 128, 1);
    gemm(bufA, 128, Wp[15], F(I_fp1 + 7), F(I_fp1 + 8), bufB, nullptr, rows, 128, 1);
    gemm(bufB, 128, Wp[16], F(I_cls1 + 1), F(I_cls1 + 2), bufA, nullptr, rows, 128, 1);
    cls2_kernel<<<(rows + 255) / 256, 256, 0, stream>>>(
        bufA, F(I_c2w), F(I_c2b), (float*)d_out, Bb, Nn);
  }
}
